// GAT_13898514170722
// MI455X (gfx1250) — compile-verified, matrix operand order checked by reading, not run on hardware
//
#include <hip/hip_runtime.h>
#include <hip/hip_bf16.h>
#include <stdint.h>

#define HID 64
#define NEG_SLOPE 0.2f
#define BN_EPS 1e-5f

typedef __attribute__((ext_vector_type(16))) __bf16 v16bf;
typedef __attribute__((ext_vector_type(8)))  float  v8f;

// ---- monotonic int encoding of float for atomicMax-based segment max ----
__device__ __forceinline__ int enc_f32(float x) {
    int b = __float_as_int(x);
    return b >= 0 ? b : (b ^ 0x7fffffff);
}
__device__ __forceinline__ float dec_f32(int e) {
    int b = e >= 0 ? e : (e ^ 0x7fffffff);
    return __int_as_float(b);
}

// ============================================================================
// Dense GEMM  out[N x 64] = A[N x K] * W[K x 64]  via bf16 WMMA, f32 accum.
// Block = 128 threads (4 waves). Block owns 16 rows; wave w owns cols [16w,16w+16).
// A-frag layout (16-bit A 16x32): lane%16 = row, lane/16 selects K octets
//   {half*8 .. half*8+7} and {16+half*8 .. 16+half*8+7}.
// B-frag mirrors with lane%16 = col.
// D: VGPR r -> row = 8*half + r, col = lane%16.
// ============================================================================
__global__ void __launch_bounds__(128)
gemm_bf16_wmma(const float* __restrict__ A, const float* __restrict__ W,
               float* __restrict__ out, int K) {
    const int lane = threadIdx.x & 31;
    const int wave = threadIdx.x >> 5;
    const int half = lane >> 4;
    const int m    = lane & 15;
    const int row  = blockIdx.x * 16 + m;       // A row held by this lane
    const int col  = wave * 16 + m;             // B/D column held by this lane

    v8f acc = {};
    for (int kc = 0; kc < K; kc += 32) {
        const float* ap = A + (size_t)row * K + kc + half * 8;
        float4 a0 = *(const float4*)(ap);
        float4 a1 = *(const float4*)(ap + 4);
        float4 a2 = *(const float4*)(ap + 16);
        float4 a3 = *(const float4*)(ap + 20);

        v16bf af, bfv;
        af[0]=(__bf16)a0.x; af[1]=(__bf16)a0.y; af[2]=(__bf16)a0.z; af[3]=(__bf16)a0.w;
        af[4]=(__bf16)a1.x; af[5]=(__bf16)a1.y; af[6]=(__bf16)a1.z; af[7]=(__bf16)a1.w;
        af[8]=(__bf16)a2.x; af[9]=(__bf16)a2.y; af[10]=(__bf16)a2.z; af[11]=(__bf16)a2.w;
        af[12]=(__bf16)a3.x; af[13]=(__bf16)a3.y; af[14]=(__bf16)a3.z; af[15]=(__bf16)a3.w;

        #pragma unroll
        for (int j = 0; j < 16; ++j) {
            int k = (j < 8) ? (kc + half * 8 + j) : (kc + 16 + half * 8 + (j - 8));
            bfv[j] = (__bf16)W[(size_t)k * HID + col];
        }

        acc = __builtin_amdgcn_wmma_f32_16x16x32_bf16(
            /*neg_a=*/false, af, /*neg_b=*/false, bfv,
            /*c_mod=*/(short)0, acc, /*reuse_a=*/false, /*reuse_b=*/false);
    }

    #pragma unroll
    for (int r = 0; r < 8; ++r) {
        int orow = blockIdx.x * 16 + half * 8 + r;
        out[(size_t)orow * HID + col] = acc[r];
    }
}

// ---- per-node attention dot products: as = h . att[0], ad = h . att[1] ----
__global__ void alpha_kernel(const float* __restrict__ hlin, const float* __restrict__ att,
                             float* __restrict__ as_, float* __restrict__ ad_, int n) {
    int i = blockIdx.x * blockDim.x + threadIdx.x;
    if (i >= n) return;
    const float* h = hlin + (size_t)i * HID;
    float s = 0.f, d = 0.f;
    #pragma unroll
    for (int k = 0; k < HID; ++k) {
        float v = h[k];
        s += v * att[k];
        d += v * att[HID + k];
    }
    as_[i] = s;
    ad_[i] = d;
}

// ---- init per-layer accumulators ----
__global__ void init_layer(float* __restrict__ agg, float* __restrict__ segsum,
                           int* __restrict__ segmax, float* __restrict__ colsum,
                           float* __restrict__ colsumsq, int n) {
    int t = blockIdx.x * blockDim.x + threadIdx.x;
    if (t < n * HID) agg[t] = 0.f;
    if (t < n) { segsum[t] = 0.f; segmax[t] = (int)0x80000000; }
    if (t < HID) { colsum[t] = 0.f; colsumsq[t] = 0.f; }
}

// ---- edge pass 1: leaky-relu logit + segment max over dst ----
__global__ void edge_max(const int* __restrict__ ei, int E, int n,
                         const float* __restrict__ as_, const float* __restrict__ ad_,
                         float* __restrict__ ea, int* __restrict__ segmax) {
    int t = blockIdx.x * blockDim.x + threadIdx.x;
    if (t >= E + n) return;
    int s = (t < E) ? ei[t]     : (t - E);   // self-loops appended
    int d = (t < E) ? ei[E + t] : (t - E);
    float a = as_[s] + ad_[d];
    a = a > 0.f ? a : NEG_SLOPE * a;
    ea[t] = a;
    atomicMax(segmax + d, enc_f32(a));
}

// ---- edge pass 2: e = exp(a - max[dst]); segment sum over dst ----
__global__ void edge_expsum(const int* __restrict__ ei, int E, int n,
                            const int* __restrict__ segmax, float* __restrict__ ea,
                            float* __restrict__ segsum) {
    int t = blockIdx.x * blockDim.x + threadIdx.x;
    if (t >= E + n) return;
    int d = (t < E) ? ei[E + t] : (t - E);
    float e = __expf(ea[t] - dec_f32(segmax[d]));
    ea[t] = e;
    atomicAdd(segsum + d, e);
}

// ---- edge pass 3: agg[dst] += (e/denom[dst]) * hlin[src]; wave per edge ----
__global__ void edge_agg(const int* __restrict__ ei, int E, int n,
                         const float* __restrict__ ea, const float* __restrict__ segsum,
                         const float* __restrict__ hlin, float* __restrict__ agg) {
    unsigned t = blockIdx.x * blockDim.x + threadIdx.x;
    unsigned edge = t >> 5;
    int lane = t & 31;
    if (edge >= (unsigned)(E + n)) return;
    int s = (edge < (unsigned)E) ? ei[edge]     : (int)(edge - E);
    int d = (edge < (unsigned)E) ? ei[E + edge] : (int)(edge - E);
    float coef = ea[edge] / segsum[d];
    const float* hs = hlin + (size_t)s * HID;
    float* od = agg + (size_t)d * HID;
    atomicAdd(od + lane,      coef * hs[lane]);
    atomicAdd(od + lane + 32, coef * hs[lane + 32]);
}

// ---- BN column statistics: partial sums per block -> global atomics ----
__global__ void __launch_bounds__(256)
bn_stats(const float* __restrict__ agg, int n,
         float* __restrict__ colsum, float* __restrict__ colsumsq) {
    __shared__ float ss[256], sq[256];
    int c  = threadIdx.x & 63;
    int rg = threadIdx.x >> 6;            // 4 row-groups per block
    float s = 0.f, q = 0.f;
    for (int r = blockIdx.x * 4 + rg; r < n; r += gridDim.x * 4) {
        float v = agg[(size_t)r * HID + c];
        s += v; q += v * v;
    }
    ss[threadIdx.x] = s; sq[threadIdx.x] = q;
    __syncthreads();
    if (threadIdx.x < 64) {
        s = ss[threadIdx.x] + ss[threadIdx.x + 64] + ss[threadIdx.x + 128] + ss[threadIdx.x + 192];
        q = sq[threadIdx.x] + sq[threadIdx.x + 64] + sq[threadIdx.x + 128] + sq[threadIdx.x + 192];
        atomicAdd(colsum + threadIdx.x, s);
        atomicAdd(colsumsq + threadIdx.x, q);
    }
}

// ---- BN apply + ReLU (GAT bias cancels inside BN, dropped exactly) ----
__global__ void bn_relu(const float* __restrict__ agg, int n,
                        const float* __restrict__ colsum, const float* __restrict__ colsumsq,
                        const float* __restrict__ gb, float* __restrict__ hout) {
    int t = blockIdx.x * blockDim.x + threadIdx.x;
    if (t >= n * HID) return;
    int c = t & 63;
    float inv_n = 1.f / (float)n;
    float mean = colsum[c] * inv_n;
    float var  = colsumsq[c] * inv_n - mean * mean;
    float y = (agg[t] - mean) * rsqrtf(var + BN_EPS) * gb[c] + gb[HID + c];
    hout[t] = y > 0.f ? y : 0.f;
}

// ---- classifier: out[N x 16] = h[N x 64] * clsW[64 x 16] + clsb ----
__global__ void classify(const float* __restrict__ h, const float* __restrict__ W,
                         const float* __restrict__ b, float* __restrict__ out, int n) {
    int t = blockIdx.x * blockDim.x + threadIdx.x;
    if (t >= n * 16) return;
    int i = t >> 4, c = t & 15;
    const float* hp = h + (size_t)i * HID;
    float s = b[c];
    #pragma unroll
    for (int k = 0; k < HID; ++k) s += hp[k] * W[k * 16 + c];
    out[t] = s;
}

extern "C" void kernel_launch(void* const* d_in, const int* in_sizes, int n_in,
                              void* d_out, int out_size, void* d_ws, size_t ws_size,
                              hipStream_t stream) {
    const float* x    = (const float*)d_in[0];
    const int*   ei   = (const int*)  d_in[1];   // [2, E] int32, src row then dst row
    const float* W0   = (const float*)d_in[2];
    const float* att0 = (const float*)d_in[3];
    const float* bn0  = (const float*)d_in[5];   // [gamma(64); beta(64)]
    const float* Wr   = (const float*)d_in[6];   // [2,64,64]
    const float* attr = (const float*)d_in[7];   // [2,2,64]
    const float* bnr  = (const float*)d_in[9];   // [2,2,64]
    const float* clsW = (const float*)d_in[10];
    const float* clsb = (const float*)d_in[11];
    (void)n_in; (void)out_size; (void)ws_size;

    const int n = in_sizes[0] / 128;   // 100000 (divisible by 16)
    const int E = in_sizes[1] / 2;     // 1600000
    const int total_e = E + n;         // with self-loops

    // ---- carve workspace (256B aligned slices) ----
    char* wp = (char*)d_ws;
    auto carve = [&](size_t bytes) -> void* {
        void* p = (void*)wp;
        wp += (bytes + 255) & ~(size_t)255;
        return p;
    };
    float* h        = (float*)carve((size_t)n * HID * 4);
    float* hlin     = (float*)carve((size_t)n * HID * 4);
    float* agg      = (float*)carve((size_t)n * HID * 4);
    float* as_      = (float*)carve((size_t)n * 4);
    float* ad_      = (float*)carve((size_t)n * 4);
    int*   segmax   = (int*)  carve((size_t)n * 4);
    float* segsum   = (float*)carve((size_t)n * 4);
    float* ea       = (float*)carve((size_t)total_e * 4);
    float* colsum   = (float*)carve(HID * 4);
    float* colsumsq = (float*)carve(HID * 4);

    auto run_layer = [&](const float* xin, int K, const float* W,
                         const float* att, const float* gb) {
        init_layer<<<(n * HID + 255) / 256, 256, 0, stream>>>(agg, segsum, segmax,
                                                              colsum, colsumsq, n);
        gemm_bf16_wmma<<<n / 16, 128, 0, stream>>>(xin, W, hlin, K);
        alpha_kernel<<<(n + 255) / 256, 256, 0, stream>>>(hlin, att, as_, ad_, n);
        edge_max<<<(total_e + 255) / 256, 256, 0, stream>>>(ei, E, n, as_, ad_, ea, segmax);
        edge_expsum<<<(total_e + 255) / 256, 256, 0, stream>>>(ei, E, n, segmax, ea, segsum);
        edge_agg<<<((unsigned)total_e * 32u + 255u) / 256u, 256, 0, stream>>>(
            ei, E, n, ea, segsum, hlin, agg);
        bn_stats<<<256, 256, 0, stream>>>(agg, n, colsum, colsumsq);
        bn_relu<<<(n * HID + 255) / 256, 256, 0, stream>>>(agg, n, colsum, colsumsq, gb, h);
    };

    run_layer(x, 128, W0, att0, bn0);                               // layer 0: 128 -> 64
    run_layer(h,  64, Wr,            attr,       bnr);              // layer 1
    run_layer(h,  64, Wr + 64 * 64,  attr + 128, bnr + 128);        // layer 2

    classify<<<((unsigned)n * 16u + 255u) / 256u, 256, 0, stream>>>(h, clsW, clsb,
                                                                    (float*)d_out, n);
}
